// GCN_85237920957105
// MI455X (gfx1250) — matrix operand order repaired and checked by reading, not compiled, not run on hardware
//
#include <hip/hip_runtime.h>
#include <math.h>

#define NN 100000
#define EE 1600000
#define HH 128     // F_IN == H == HG == 128
#define LL 3
#define CC 40
#define CPAD 48

typedef __attribute__((ext_vector_type(16))) __bf16 bf16x16;
typedef __attribute__((ext_vector_type(8)))  __bf16 bf16x8;
typedef __attribute__((ext_vector_type(8)))  float  floatx8;

// ---------- helpers ----------
__device__ inline __bf16 f2bf(float f) {
    union { float f; unsigned u; } a; a.f = f;
    unsigned r = a.u + 0x7FFFu + ((a.u >> 16) & 1u);   // round-to-nearest-even
    union { unsigned short s; __bf16 b; } o; o.s = (unsigned short)(r >> 16);
    return o.b;
}

__device__ inline floatx8 wmma_bf16(bf16x16 a, bf16x16 b, floatx8 c) {
    return __builtin_amdgcn_wmma_f32_16x16x32_bf16(false, a, false, b,
                                                   (short)0, c, false, false);
}

// A-fragment: 16x32 bf16 tile, row-major source, ld = 128.
// lane: row = lane&15, half = lane>>4; elems 0..7 -> K = k+half*8.., 8..15 -> K = k+16+half*8..
__device__ inline bf16x16 load_a(const __bf16* __restrict__ arow, int k, int half) {
    union { bf16x16 v; bf16x8 h[2]; } a;
    a.h[0] = *(const bf16x8*)(arow + k + half * 8);
    a.h[1] = *(const bf16x8*)(arow + k + 16 + half * 8);
    return a.v;
}

// B-fragment: 32x16 bf16 tile from TRANSPOSED weights Bt[n][k], ld = 128.
// lane: n = lane&15, khalf = lane>>4; elems 0..7 -> K = k+khalf*8.., 8..15 -> K = k+16+khalf*8..
__device__ inline bf16x16 load_b(const __bf16* __restrict__ bt, int col, int k, int khalf) {
    union { bf16x16 v; bf16x8 h[2]; } b;
    b.h[0] = *(const bf16x8*)(bt + (size_t)col * HH + k + khalf * 8);
    b.h[1] = *(const bf16x8*)(bt + (size_t)col * HH + k + 16 + khalf * 8);
    return b.v;
}

// ---------- small prep kernels ----------
__global__ void k_f2b(const float* __restrict__ src, __bf16* __restrict__ dst, int n) {
    int i = blockIdx.x * blockDim.x + threadIdx.x;
    if (i < n) dst[i] = f2bf(src[i]);
}

// src[rows][cols] (row-major) -> dst[colsPad][rows] (transposed bf16, zero-padded cols)
__global__ void k_transpose_f2b(const float* __restrict__ src, __bf16* __restrict__ dst,
                                int rows, int cols, int colsPad) {
    int i = blockIdx.x * blockDim.x + threadIdx.x;
    int total = colsPad * rows;
    if (i >= total) return;
    int c = i / rows, r = i - c * rows;
    float v = (c < cols) ? src[(size_t)r * cols + c] : 0.0f;
    dst[i] = f2bf(v);
}

__global__ void k_init_h(const float* __restrict__ h0, float* __restrict__ hf,
                         __bf16* __restrict__ hb, int n) {
    int i = blockIdx.x * blockDim.x + threadIdx.x;
    if (i < n) { float v = h0[i]; hf[i] = v; hb[i] = f2bf(v); }
}

__global__ void k_zero(float* __restrict__ p, int n) {
    int i = blockIdx.x * blockDim.x + threadIdx.x;
    if (i < n) p[i] = 0.0f;
}

__global__ void k_deg(const int* __restrict__ ei, const float* __restrict__ ew,
                      float* __restrict__ deg) {
    int e = blockIdx.x * blockDim.x + threadIdx.x;
    if (e >= EE + NN) return;
    int d; float w;
    if (e < EE) { d = ei[EE + e]; w = ew[e]; } else { d = e - EE; w = 1.0f; }
    atomicAdd(deg + d, w);
}

__global__ void k_dinv(float* __restrict__ deg) {
    int i = blockIdx.x * blockDim.x + threadIdx.x;
    if (i < NN) { float v = deg[i]; deg[i] = (v > 0.0f) ? rsqrtf(v) : 0.0f; }
}

// ---------- WMMA GEMM kernels ----------
// xh = relu(x @ linW + b), bf16 output. grid = 6250 blocks x 256 thr (8 waves -> 8 col tiles)
__global__ __launch_bounds__(256) void k_linear(const __bf16* __restrict__ xb,
                                                const __bf16* __restrict__ Wt,
                                                const float* __restrict__ bias,
                                                __bf16* __restrict__ outb) {
    int lane = threadIdx.x & 31, wave = threadIdx.x >> 5;
    int rowBase = blockIdx.x * 16, colBase = wave * 16;
    int rn = lane & 15, half = lane >> 4;
    const __bf16* arow = xb + (size_t)(rowBase + rn) * HH;
    floatx8 acc = {};
    for (int k = 0; k < HH; k += 32) {
        bf16x16 a = load_a(arow, k, half);
        bf16x16 b = load_b(Wt, colBase + rn, k, half);
        acc = wmma_bf16(a, b, acc);
    }
    float bv = bias[colBase + rn];
    for (int i = 0; i < 8; ++i) {
        float v = fmaxf(acc[i] + bv, 0.0f);
        outb[(size_t)(rowBase + 8 * half + i) * HH + colBase + rn] = f2bf(v);
    }
}

// xw = xh @ convW[l] (fp32 out, no epilogue)
__global__ __launch_bounds__(256) void k_conv_gemm(const __bf16* __restrict__ xb,
                                                   const __bf16* __restrict__ Wt,
                                                   float* __restrict__ outf) {
    int lane = threadIdx.x & 31, wave = threadIdx.x >> 5;
    int rowBase = blockIdx.x * 16, colBase = wave * 16;
    int rn = lane & 15, half = lane >> 4;
    const __bf16* arow = xb + (size_t)(rowBase + rn) * HH;
    floatx8 acc = {};
    for (int k = 0; k < HH; k += 32) {
        bf16x16 a = load_a(arow, k, half);
        bf16x16 b = load_b(Wt, colBase + rn, k, half);
        acc = wmma_bf16(a, b, acc);
    }
    for (int i = 0; i < 8; ++i)
        outf[(size_t)(rowBase + 8 * half + i) * HH + colBase + rn] = acc[i];
}

// one wave per edge; float4 gather, 4 atomic f32 scatter-adds per lane
__global__ void k_scatter(const int* __restrict__ ei, const float* __restrict__ ew,
                          const float* __restrict__ dinv, const float* __restrict__ xw,
                          float* __restrict__ agg) {
    long long gid = (long long)blockIdx.x * blockDim.x + threadIdx.x;
    int e = (int)(gid >> 5);
    int lane = (int)(gid & 31);
    if (e >= EE + NN) return;
    int s, d; float w;
    if (e < EE) { s = ei[e]; d = ei[EE + e]; w = ew[e]; }
    else        { s = d = e - EE; w = 1.0f; }
    float nrm = dinv[s] * w * dinv[d];
    float4 v = ((const float4*)(xw + (size_t)s * HH))[lane];
    float* od = agg + (size_t)d * HH + lane * 4;
    atomicAdd(od + 0, v.x * nrm);
    atomicAdd(od + 1, v.y * nrm);
    atomicAdd(od + 2, v.z * nrm);
    atomicAdd(od + 3, v.w * nrm);
}

// xh = relu(agg + convB[l]) -> bf16
__global__ void k_agg_epi(const float* __restrict__ agg, const float* __restrict__ bias,
                          __bf16* __restrict__ xb) {
    int i = blockIdx.x * blockDim.x + threadIdx.x;
    if (i >= NN * HH) return;
    int col = i & (HH - 1);
    xb[i] = f2bf(fmaxf(agg[i] + bias[col], 0.0f));
}

// fused GRU cell: gi = xh@Wih^T, gh = h@Whh^T, gates, in-place h update (fp32 + bf16)
__global__ __launch_bounds__(256) void k_gru(const __bf16* __restrict__ xb,
                                             __bf16* __restrict__ hb, float* __restrict__ hf,
                                             const __bf16* __restrict__ Wih,
                                             const __bf16* __restrict__ Whh,
                                             const float* __restrict__ bih,
                                             const float* __restrict__ bhh) {
    int lane = threadIdx.x & 31, wave = threadIdx.x >> 5;
    int rowBase = blockIdx.x * 16, j = wave * 16;   // gate-local col tile
    int rn = lane & 15, half = lane >> 4;
    const __bf16* ax = xb + (size_t)(rowBase + rn) * HH;
    const __bf16* ah = hb + (size_t)(rowBase + rn) * HH;
    floatx8 ir = {}, iz = {}, inn = {}, hr = {}, hz = {}, hn = {};
    for (int k = 0; k < HH; k += 32) {
        bf16x16 a_x = load_a(ax, k, half);
        bf16x16 a_h = load_a(ah, k, half);
        bf16x16 b0 = load_b(Wih,       j + rn, k, half);
        bf16x16 b1 = load_b(Wih, 128 + j + rn, k, half);
        bf16x16 b2 = load_b(Wih, 256 + j + rn, k, half);
        ir  = wmma_bf16(a_x, b0, ir);
        iz  = wmma_bf16(a_x, b1, iz);
        inn = wmma_bf16(a_x, b2, inn);
        b0 = load_b(Whh,       j + rn, k, half);
        b1 = load_b(Whh, 128 + j + rn, k, half);
        b2 = load_b(Whh, 256 + j + rn, k, half);
        hr = wmma_bf16(a_h, b0, hr);
        hz = wmma_bf16(a_h, b1, hz);
        hn = wmma_bf16(a_h, b2, hn);
    }
    __syncthreads();   // all waves done reading h_bf rows before any wave overwrites them
    int col = j + rn;
    float bir = bih[col], biz = bih[128 + col], bin = bih[256 + col];
    float bhr = bhh[col], bhz = bhh[128 + col], bhn = bhh[256 + col];
    for (int i = 0; i < 8; ++i) {
        size_t o = (size_t)(rowBase + 8 * half + i) * HH + col;
        float hold = hf[o];
        float rg = 1.0f / (1.0f + __expf(-(ir[i] + bir + hr[i] + bhr)));
        float zg = 1.0f / (1.0f + __expf(-(iz[i] + biz + hz[i] + bhz)));
        float ng = tanhf(inn[i] + bin + rg * (hn[i] + bhn));
        float hv = (1.0f - zg) * ng + zg * hold;
        hf[o] = hv;
        hb[o] = f2bf(hv);
    }
}

// out = h @ fcW + fcB ; N=40 padded to 48 (3 waves/block)
__global__ __launch_bounds__(96) void k_fc(const __bf16* __restrict__ hb,
                                           const __bf16* __restrict__ Wt,
                                           const float* __restrict__ bias,
                                           float* __restrict__ out) {
    int lane = threadIdx.x & 31, wave = threadIdx.x >> 5;
    int rowBase = blockIdx.x * 16, colBase = wave * 16;
    int rn = lane & 15, half = lane >> 4;
    const __bf16* arow = hb + (size_t)(rowBase + rn) * HH;
    floatx8 acc = {};
    for (int k = 0; k < HH; k += 32) {
        bf16x16 a = load_a(arow, k, half);
        bf16x16 b = load_b(Wt, colBase + rn, k, half);
        acc = wmma_bf16(a, b, acc);
    }
    int col = colBase + rn;
    if (col < CC) {
        float bv = bias[col];
        for (int i = 0; i < 8; ++i)
            out[(size_t)(rowBase + 8 * half + i) * CC + col] = acc[i] + bv;
    }
}

// ---------- host launch ----------
extern "C" void kernel_launch(void* const* d_in, const int* in_sizes, int n_in,
                              void* d_out, int out_size, void* d_ws, size_t ws_size,
                              hipStream_t stream) {
    const float* x    = (const float*)d_in[0];
    const int*   ei   = (const int*)  d_in[1];
    const float* ew   = (const float*)d_in[2];
    const float* h0   = (const float*)d_in[3];
    const float* linW = (const float*)d_in[4];
    const float* linB = (const float*)d_in[5];
    const float* convW= (const float*)d_in[6];
    const float* convB= (const float*)d_in[7];
    const float* Wih  = (const float*)d_in[8];
    const float* Whh  = (const float*)d_in[9];
    const float* bih  = (const float*)d_in[10];
    const float* bhh  = (const float*)d_in[11];
    const float* fcW  = (const float*)d_in[12];
    const float* fcB  = (const float*)d_in[13];
    float* out = (float*)d_out;

    char* p = (char*)d_ws;
    auto carve = [&](size_t bytes) {
        char* r = p; p += (bytes + 255) & ~(size_t)255; return r;
    };
    __bf16* x_bf   = (__bf16*)carve((size_t)NN * HH * 2);
    __bf16* xh_bf  = (__bf16*)carve((size_t)NN * HH * 2);
    __bf16* h_bf   = (__bf16*)carve((size_t)NN * HH * 2);
    float*  h_f    = (float*) carve((size_t)NN * HH * 4);
    float*  xw     = (float*) carve((size_t)NN * HH * 4);
    float*  agg    = (float*) carve((size_t)NN * HH * 4);
    float*  dinv   = (float*) carve((size_t)NN * 4);
    __bf16* linWt  = (__bf16*)carve((size_t)HH * HH * 2);
    __bf16* convWt = (__bf16*)carve((size_t)LL * HH * HH * 2);
    __bf16* WihB   = (__bf16*)carve((size_t)3 * HH * HH * 2);
    __bf16* WhhB   = (__bf16*)carve((size_t)3 * HH * HH * 2);
    __bf16* fcWt   = (__bf16*)carve((size_t)CPAD * HH * 2);

    const int T = 256;
    // weight conversion (GRU weights already in [out][k] = transposed layout)
    k_f2b<<<(3 * HH * HH + T - 1) / T, T, 0, stream>>>(Wih, WihB, 3 * HH * HH);
    k_f2b<<<(3 * HH * HH + T - 1) / T, T, 0, stream>>>(Whh, WhhB, 3 * HH * HH);
    k_transpose_f2b<<<(HH * HH + T - 1) / T, T, 0, stream>>>(linW, linWt, HH, HH, HH);
    for (int l = 0; l < LL; ++l)
        k_transpose_f2b<<<(HH * HH + T - 1) / T, T, 0, stream>>>(
            convW + (size_t)l * HH * HH, convWt + (size_t)l * HH * HH, HH, HH, HH);
    k_transpose_f2b<<<(CPAD * HH + T - 1) / T, T, 0, stream>>>(fcW, fcWt, HH, CC, CPAD);

    // activations / state
    k_f2b<<<(NN * HH + T - 1) / T, T, 0, stream>>>(x, x_bf, NN * HH);
    k_init_h<<<(NN * HH + T - 1) / T, T, 0, stream>>>(h0, h_f, h_bf, NN * HH);

    // degree -> dinv (self-loops included)
    k_zero<<<(NN + T - 1) / T, T, 0, stream>>>(dinv, NN);
    k_deg<<<(EE + NN + T - 1) / T, T, 0, stream>>>(ei, ew, dinv);
    k_dinv<<<(NN + T - 1) / T, T, 0, stream>>>(dinv);

    const int MB = NN / 16;  // 6250 row-tiles, exact
    k_linear<<<MB, 256, 0, stream>>>(x_bf, linWt, linB, xh_bf);
    k_gru<<<MB, 256, 0, stream>>>(xh_bf, h_bf, h_f, WihB, WhhB, bih, bhh);

    const int SCB = (int)(((long long)(EE + NN) * 32 + T - 1) / T);
    for (int l = 0; l < LL; ++l) {
        k_conv_gemm<<<MB, 256, 0, stream>>>(xh_bf, convWt + (size_t)l * HH * HH, xw);
        k_zero<<<(NN * HH + T - 1) / T, T, 0, stream>>>(agg, NN * HH);
        k_scatter<<<SCB, T, 0, stream>>>(ei, ew, dinv, xw, agg);
        k_agg_epi<<<(NN * HH + T - 1) / T, T, 0, stream>>>(agg, convB + (size_t)l * HH, xh_bf);
        k_gru<<<MB, 256, 0, stream>>>(xh_bf, h_bf, h_f, WihB, WhhB, bih, bhh);
    }
    k_fc<<<MB, 96, 0, stream>>>(h_bf, fcWt, fcB, out);
}
